// FP8PerChannelLinear_30777735643613
// MI455X (gfx1250) — matrix operand order repaired and checked by reading, not compiled, not run on hardware
//
#include <hip/hip_runtime.h>
#include <hip/hip_bf16.h>

typedef __attribute__((ext_vector_type(16))) _Float16 v16h;
typedef __attribute__((ext_vector_type(8)))  _Float16 v8h;
typedef __attribute__((ext_vector_type(4)))  _Float16 v4h;
typedef __attribute__((ext_vector_type(8)))  float    v8f;
typedef __attribute__((ext_vector_type(4)))  float    v4f;

#define FP8_MAX 448.0f
#define GK 1024   // in_features
#define GN 1024   // out_features
#define TM 128    // block tile M
#define TNB 128   // block tile N
#define TK 64     // K halves staged per iteration (two 32-K WMMA steps)
#define PITCH 72  // LDS row pitch in halves (144B = 36 banks; gcd(36,64)=4 -> conflict-free)

// LDS byte offset of a __shared__ object (ptrtoint of addrspace(3) pointer)
static __device__ __forceinline__ unsigned lds_addr(const void* p) {
    return (unsigned)(unsigned long long)(const __attribute__((address_space(3))) void*)p;
}

// CDNA5 async global->LDS (ASYNCcnt-tracked, no VGPR data movement)
#define ASYNC_LD_B128(ldsoff, gptr)                                              \
    asm volatile("global_load_async_to_lds_b128 %0, %1, off"                     \
                 :: "v"(ldsoff), "v"(gptr) : "memory")
#define WAIT_ASYNC0() asm volatile("s_wait_asynccnt 0x0" ::: "memory")

// ---------------- Kernel 1: per-tensor amax of x (uint-bit atomic max) ----------------
__global__ __launch_bounds__(256) void fp8lin_amax_kernel(const float* __restrict__ x,
                                                          unsigned* __restrict__ amax_bits,
                                                          int n) {
    int n4 = n >> 2;
    float m = 0.0f;
    const v4f* x4 = (const v4f*)x;
    for (int i = blockIdx.x * blockDim.x + threadIdx.x; i < n4; i += gridDim.x * blockDim.x) {
        v4f v = x4[i];
        m = fmaxf(m, fmaxf(fmaxf(fabsf(v[0]), fabsf(v[1])), fmaxf(fabsf(v[2]), fabsf(v[3]))));
    }
    __shared__ float red[256];
    red[threadIdx.x] = m;
    __syncthreads();
    for (int s = 128; s > 0; s >>= 1) {
        if (threadIdx.x < (unsigned)s) red[threadIdx.x] = fmaxf(red[threadIdx.x], red[threadIdx.x + s]);
        __syncthreads();
    }
    if (threadIdx.x == 0) atomicMax(amax_bits, __float_as_uint(red[0]));
}

// ------- Kernel 2: per-output-row weight amax, scale, quantize weight to fp16 -------
__global__ __launch_bounds__(256) void fp8lin_wq_kernel(const float* __restrict__ w,
                                                        float* __restrict__ wscale,
                                                        _Float16* __restrict__ wq,
                                                        const unsigned* __restrict__ amax_bits,
                                                        float* __restrict__ in_scale) {
    int o = blockIdx.x;
    int tid = threadIdx.x;
    const float* row = w + (size_t)o * GK;
    float m = 0.0f;
    for (int i = tid; i < GK; i += 256) m = fmaxf(m, fabsf(row[i]));
    __shared__ float red[256];
    red[tid] = m;
    __syncthreads();
    for (int s = 128; s > 0; s >>= 1) {
        if (tid < (unsigned)s) red[tid] = fmaxf(red[tid], red[tid + s]);
        __syncthreads();
    }
    __shared__ float ssc;
    if (tid == 0) {
        float sc = fmaxf(red[0] / FP8_MAX, 1e-12f);
        wscale[o] = sc;
        ssc = sc;
        if (o == 0)  // amax kernel completed earlier on same stream
            *in_scale = fmaxf(__uint_as_float(*amax_bits) / FP8_MAX, 1e-12f);
    }
    __syncthreads();
    float sc = ssc;
    for (int i = tid; i < GK; i += 256) {
        float q = rintf(fminf(fmaxf(row[i] / sc, -FP8_MAX), FP8_MAX));  // RNE like jnp.round
        wq[(size_t)o * GK + i] = (_Float16)q;  // integers <=448 exact in fp16
    }
}

// ---------------- Kernel 3: quantize x to integer-valued fp16 ----------------
__global__ __launch_bounds__(256) void fp8lin_xq_kernel(const float* __restrict__ x,
                                                        _Float16* __restrict__ xq,
                                                        const float* __restrict__ in_scale,
                                                        int n) {
    float inv = 1.0f / *in_scale;
    int n4 = n >> 2;
    const v4f* x4 = (const v4f*)x;
    v4h* q4 = (v4h*)xq;
    for (int i = blockIdx.x * blockDim.x + threadIdx.x; i < n4; i += gridDim.x * blockDim.x) {
        v4f v = x4[i];
        v4h q;
#pragma unroll
        for (int j = 0; j < 4; ++j)
            q[j] = (_Float16)rintf(fminf(fmaxf(v[j] * inv, -FP8_MAX), FP8_MAX));
        q4[i] = q;
    }
}

// ---------------- WMMA fragment gathers (ISA 7.12.2 layouts) ----------------
// A 16x32 f16 frag: lanes<16 row=l16 K{0..7,16..23}; lanes>=16 row=l16 K{8..15,24..31}
static __device__ __forceinline__ v16h load_fragA(const _Float16* buf, int row, int kk, int hi) {
    const _Float16* p = buf + row * PITCH + kk;
    v8h lo = *(const v8h*)(p + hi * 8);
    v8h hh = *(const v8h*)(p + 16 + hi * 8);
    return __builtin_shufflevector(lo, hh, 0, 1, 2, 3, 4, 5, 6, 7, 8, 9, 10, 11, 12, 13, 14, 15);
}
// B 32x16 f16 frag: lanes<16 col=l16 K{0..7,16..23}; lanes>=16 col=l16 K{8..15,24..31}
static __device__ __forceinline__ v16h load_fragB(const _Float16* buf, int col, int kk, int hi) {
    const _Float16* p = buf + col * PITCH + kk;
    v8h lo = *(const v8h*)(p + hi * 8);
    v8h hh = *(const v8h*)(p + 16 + hi * 8);
    return __builtin_shufflevector(lo, hh, 0, 1, 2, 3, 4, 5, 6, 7, 8, 9, 10, 11, 12, 13, 14, 15);
}

// ---- Kernel 4: double-buffered async-LDS WMMA GEMM + dequant + bias ----
// out[t,o] = (sum_k xq[t,k]*wq[o,k]) * in_scale*wscale[o] + bias[o]
__global__ __launch_bounds__(256) void fp8lin_gemm_kernel(const _Float16* __restrict__ Aq,
                                                          const _Float16* __restrict__ Wq,
                                                          const float* __restrict__ wscale,
                                                          const float* __restrict__ in_scale,
                                                          const float* __restrict__ bias,
                                                          float* __restrict__ out) {
    __shared__ _Float16 sA[2][TM * PITCH];   // 2 x 18432 B
    __shared__ _Float16 sB[2][TNB * PITCH];  // 2 x 18432 B

    const int tid = threadIdx.x;
    const int wave = tid >> 5;
    const int lane = tid & 31;
    const int l16 = lane & 15;
    const int hi = lane >> 4;

    const int m_blk = blockIdx.x * TM;
    const int n_blk = blockIdx.y * TNB;
    const int wm = (wave >> 1) * 32;  // 0,32,64,96
    const int wn = (wave & 1) * 64;   // 0,64

    // staging: 8 chunks (16B) per row of 64 halves; 1024 chunks per 128-row tile
    const int r0 = tid >> 3;          // 0..31 (then +32,+64,+96)
    const int off = (tid & 7) << 3;   // 0..56 halves
    const _Float16* gA0 = Aq + (size_t)(m_blk + r0) * GK + off;
    const _Float16* gB0 = Wq + (size_t)(n_blk + r0) * GK + off;

    unsigned laA[2], laB[2];
    laA[0] = lds_addr(&sA[0][r0 * PITCH + off]);
    laA[1] = lds_addr(&sA[1][r0 * PITCH + off]);
    laB[0] = lds_addr(&sB[0][r0 * PITCH + off]);
    laB[1] = lds_addr(&sB[1][r0 * PITCH + off]);
    const unsigned rowblk = 32 * PITCH * sizeof(_Float16);  // 32 rows of LDS, bytes

    v8f acc[2][4] = {};

    // prologue: stage iteration 0 into buffer 0
#pragma unroll
    for (int c = 0; c < 4; ++c) {
        ASYNC_LD_B128(laA[0] + c * rowblk, gA0 + (size_t)c * 32 * GK);
        ASYNC_LD_B128(laB[0] + c * rowblk, gB0 + (size_t)c * 32 * GK);
    }
    WAIT_ASYNC0();
    __syncthreads();

    const int NIT = GK / TK;  // 16
    for (int it = 0; it < NIT; ++it) {
        const int cur = it & 1;
        // issue async loads for next iteration into the other buffer
        if (it + 1 < NIT) {
            const int k0 = (it + 1) * TK;
            const int nxt = cur ^ 1;
#pragma unroll
            for (int c = 0; c < 4; ++c) {
                ASYNC_LD_B128(laA[nxt] + c * rowblk, gA0 + k0 + (size_t)c * 32 * GK);
                ASYNC_LD_B128(laB[nxt] + c * rowblk, gB0 + k0 + (size_t)c * 32 * GK);
            }
        }
        // compute on current buffer: 2 K-steps x (2 A-frags x 4 B-frags) = 16 WMMA
        const _Float16* bA = &sA[cur][0];
        const _Float16* bB = &sB[cur][0];
#pragma unroll
        for (int kk = 0; kk < TK; kk += 32) {
            v16h a0 = load_fragA(bA, wm + l16, kk, hi);
            v16h a1 = load_fragA(bA, wm + 16 + l16, kk, hi);
#pragma unroll
            for (int j = 0; j < 4; ++j) {
                v16h b = load_fragB(bB, wn + j * 16 + l16, kk, hi);
                acc[0][j] = __builtin_amdgcn_wmma_f32_16x16x32_f16(false, a0, false, b, (short)0, acc[0][j], false, false);
                acc[1][j] = __builtin_amdgcn_wmma_f32_16x16x32_f16(false, a1, false, b, (short)0, acc[1][j], false, false);
            }
        }
        // next buffer ready + all waves done reading current buffer
        WAIT_ASYNC0();
        __syncthreads();
    }

    // Epilogue: C/D layout — lanes<16: N=l16, M=vgpr; lanes>=16: N=l16, M=8+vgpr
    const float isc = *in_scale;
#pragma unroll
    for (int mi = 0; mi < 2; ++mi) {
#pragma unroll
        for (int nj = 0; nj < 4; ++nj) {
            int col = n_blk + wn + nj * 16 + l16;
            float sc = isc * wscale[col];
            float bs = bias[col];
            int re = m_blk + wm + mi * 16 + hi * 8;
#pragma unroll
            for (int v = 0; v < 8; ++v)
                out[(size_t)(re + v) * GN + col] = acc[mi][nj][v] * sc + bs;
        }
    }
}

extern "C" void kernel_launch(void* const* d_in, const int* in_sizes, int n_in,
                              void* d_out, int out_size, void* d_ws, size_t ws_size,
                              hipStream_t stream) {
    const float* x    = (const float*)d_in[0];
    const float* w    = (const float*)d_in[1];
    const float* bias = (const float*)d_in[2];
    float* out = (float*)d_out;

    const int n_x = in_sizes[0];  // M*K
    const int M = n_x / GK;       // 32768

    char* ws = (char*)d_ws;
    unsigned* amax_bits = (unsigned*)ws;           // 4 B
    float* in_scale     = (float*)(ws + 8);        // 4 B
    float* wscale       = (float*)(ws + 64);       // 4 KB
    _Float16* wq        = (_Float16*)(ws + 8192);  // 2 MB
    _Float16* xq        = (_Float16*)(ws + 8192 + (size_t)GN * GK * sizeof(_Float16));  // 64 MB

    hipMemsetAsync(amax_bits, 0, sizeof(unsigned), stream);

    fp8lin_amax_kernel<<<1024, 256, 0, stream>>>(x, amax_bits, n_x);
    fp8lin_wq_kernel<<<GN, 256, 0, stream>>>(w, wscale, wq, amax_bits, in_scale);
    fp8lin_xq_kernel<<<2048, 256, 0, stream>>>(x, xq, in_scale, n_x);

    dim3 grid(M / TM, GN / TNB);  // 256 x 8
    fp8lin_gemm_kernel<<<grid, 256, 0, stream>>>(xq, wq, wscale, in_scale, bias, out);
}
